// BiQRNNAtt_8186207666223
// MI455X (gfx1250) — matrix operand order repaired and checked by reading, not compiled, not run on hardware
//
#include <hip/hip_runtime.h>

// ---- problem constants ----
#define BQ_B   64
#define BQ_S   2048
#define BQ_M   (BQ_B * BQ_S)   // 131072 rows
#define BQ_KP  160             // 132 padded to multiple of 32
#define BQ_N   1024            // [Zf 0:256 | Ff 256:512 | Zb 512:768 | Fb 768:1024]
#define BQ_H   256

typedef unsigned int   u32;
typedef unsigned short u16;
typedef __attribute__((ext_vector_type(16))) __bf16 v16bf;
typedef __attribute__((ext_vector_type(8)))  float  v8f;

__device__ __forceinline__ u16 f2bf(float f) {
    u32 u = __float_as_uint(f);
    u32 r = u + 0x7FFFu + ((u >> 16) & 1u);   // round-to-nearest-even
    return (u16)(r >> 16);
}
__device__ __forceinline__ float bf2f(u16 h) {
    return __uint_as_float(((u32)h) << 16);
}

// ---------------------------------------------------------------------------
// Kernel 1: pack combined weight matrix Wc (KP x 1024, bf16) + bias (1024 f32)
// Wc[:,0:512]  = Wf[:,0:512] (Z,F cols; O cols unused by reference)
// Wc[:,512:]   = Wb[:,0:512] ; rows 132..159 zero padding
// ---------------------------------------------------------------------------
__global__ void bq_prep_w(const float* __restrict__ Wf, const float* __restrict__ Wb,
                          const float* __restrict__ bfv, const float* __restrict__ bbv,
                          u16* __restrict__ Wc, float* __restrict__ bias) {
    int i = blockIdx.x * 256 + threadIdx.x;
    if (i < BQ_KP * BQ_N) {
        int k = i >> 10, n = i & 1023;
        float v = 0.f;
        if (k < 132) v = (n < 512) ? Wf[k * 768 + n] : Wb[k * 768 + (n - 512)];
        Wc[i] = f2bf(v);
    }
    if (i < BQ_N) bias[i] = (i < 512) ? bfv[i] : bbv[i - 512];
}

// ---------------------------------------------------------------------------
// Kernel 2: build x (M x KP, bf16): [emb[ev] (128) | num@num_W+num_b (4) | pad]
// One wave per row; lane covers 5 columns.
// ---------------------------------------------------------------------------
__global__ void bq_build_x(const float* __restrict__ X, const float* __restrict__ emb,
                           const float* __restrict__ nW, const float* __restrict__ nb,
                           u16* __restrict__ xq) {
    int row  = blockIdx.x * 8 + (threadIdx.x >> 5);
    int lane = threadIdx.x & 31;
    const float* xr = X + (size_t)row * 8;
    int ev = (int)xr[0];
    float nums[7];
#pragma unroll
    for (int i = 0; i < 7; i++) nums[i] = xr[1 + i];
    u16* dst = xq + (size_t)row * BQ_KP;
#pragma unroll
    for (int kk = 0; kk < 5; kk++) {
        int col = lane + kk * 32;
        float v;
        if (col < 128) {
            v = emb[ev * 128 + col];
        } else if (col < 132) {
            int j = col - 128;
            float s = nb[j];
#pragma unroll
            for (int i = 0; i < 7; i++) s += nums[i] * nW[i * 4 + j];
            v = s;
        } else {
            v = 0.f;
        }
        dst[col] = f2bf(v);
    }
}

// ---------------------------------------------------------------------------
// Kernel 3: WMMA GEMM  G = act(x @ Wc + bias)   (M x 1024, bf16 out)
// Block: 256 thr = 8 waves; tile 128M x 64N; K = 160 in 5 chunks of 32.
// Fused epilogue: tanh on Z zones (cg>>8 even), sigmoid on F zones (odd).
// ---------------------------------------------------------------------------
__global__ __launch_bounds__(256) void bq_gemm_gates(const u16* __restrict__ xq,
                                                     const u16* __restrict__ Wc,
                                                     const float* __restrict__ bias,
                                                     u16* __restrict__ G) {
    __shared__ u16 As[128 * BQ_KP];   // [m][k], rows contiguous (matches global)
    __shared__ u16 Bs[64 * BQ_KP];    // transposed: [n][k] so K-pairs are dwords

    const int tid    = threadIdx.x;
    const int blockM = blockIdx.x * 128;
    const int n0     = blockIdx.y * 64;

    // Stage A: 128 rows * 160 cols bf16 = contiguous 40960 B => 2560 uint4
    {
        const uint4* src = (const uint4*)(xq + (size_t)blockM * BQ_KP);
        uint4* dst = (uint4*)As;
#pragma unroll
        for (int i = 0; i < 10; i++) dst[tid + i * 256] = src[tid + i * 256];
    }
    // Stage B transposed: coalesced global reads, scattered LDS writes
    for (int e = tid; e < 64 * BQ_KP; e += 256) {
        int col = e & 63;
        int k   = e >> 6;
        Bs[col * BQ_KP + k] = Wc[(size_t)k * BQ_N + n0 + col];
    }
    __syncthreads();

    const int w    = tid >> 5;     // wave id 0..7 -> M strip
    const int lane = tid & 31;
    const int ml   = lane & 15;
    const int half = lane >> 4;

    v8f acc[4] = {v8f{}, v8f{}, v8f{}, v8f{}};
    const u16* arow = As + (w * 16 + ml) * BQ_KP;

#pragma unroll
    for (int kc = 0; kc < 5; kc++) {
        const int kb = kc * 32;
        // A fragment (16x32 bf16): VGPR v<4: K = half*8 + 2v ; v>=4: K = 16 + half*8 + 2(v-4)
        union { u32 u[8]; v16bf v; } au;
#pragma unroll
        for (int v2 = 0; v2 < 8; v2++) {
            int k0 = ((v2 < 4) ? 0 : 16) + half * 8 + 2 * (v2 & 3);
            au.u[v2] = *(const u32*)(arow + kb + k0);
        }
#pragma unroll
        for (int j = 0; j < 4; j++) {
            // B fragment (32x16 bf16): N = lane&15 (per 16-tile), VGPR v<4: K = half*8 + 2v ; v>=4: K = 16 + half*8 + 2(v-4)
            const u16* bj = Bs + (j * 16 + ml) * BQ_KP;
            union { u32 u[8]; v16bf v; } bu;
#pragma unroll
            for (int v2 = 0; v2 < 8; v2++) {
                bu.u[v2] = *(const u32*)(bj + kb + ((v2 < 4) ? 0 : 16) + half * 8 + 2 * (v2 & 3));
            }
            acc[j] = __builtin_amdgcn_wmma_f32_16x16x32_bf16(
                false, au.v, false, bu.v, (short)0, acc[j], false, false);
        }
    }

    // Epilogue: D layout — lane<16: M=r ; lane>=16: M=8+r ; N = lane&15
#pragma unroll
    for (int j = 0; j < 4; j++) {
        int cg   = n0 + j * 16 + ml;
        float bs = bias[cg];
        bool sig = (cg >> 8) & 1;     // zones 1,3 (F gates) -> sigmoid; 0,2 -> tanh
#pragma unroll
        for (int r = 0; r < 8; r++) {
            int   row = blockM + w * 16 + half * 8 + r;
            float val = acc[j][r] + bs;
            val = sig ? (1.f / (1.f + __expf(-val))) : tanhf(val);
            G[(size_t)row * BQ_N + cg] = f2bf(val);
        }
    }
}

// ---------------------------------------------------------------------------
// Kernel 4: fo-pool recurrences. 32768 threads = 2 dirs x 64 b x 256 h.
// h_bwd follows the reference: pool over time-reversed gates, stored un-reversed.
// ---------------------------------------------------------------------------
__global__ void bq_fo_pool(const u16* __restrict__ G, u16* __restrict__ hc) {
    int gid = blockIdx.x * 256 + threadIdx.x;
    int dir = gid >> 14;
    int r   = gid & 16383;
    int b   = r >> 8;
    int h   = r & 255;
    float hs = 0.f;
    const size_t base = (size_t)b * BQ_S;
    if (dir == 0) {
        for (int t = 0; t < BQ_S; t++) {
            size_t g = (base + t) * BQ_N;
            float z = bf2f(G[g + h]);
            float f = bf2f(G[g + 256 + h]);
            hs = f * hs + (1.f - f) * z;
            hc[(base + t) * 512 + h] = f2bf(hs);
        }
    } else {
        for (int i = 0; i < BQ_S; i++) {
            size_t g = (base + (BQ_S - 1 - i)) * BQ_N;
            float z = bf2f(G[g + 512 + h]);
            float f = bf2f(G[g + 768 + h]);
            hs = f * hs + (1.f - f) * z;
            hc[(base + i) * 512 + 256 + h] = f2bf(hs);
        }
    }
}

// ---------------------------------------------------------------------------
// Kernel 5: attention + output. One block (512 thr = 16 waves) per batch.
// scores = hc @ Mu ; softmax over S ; ctx = sum_t w_t * hc ; out = ctx@out_W+b
// ---------------------------------------------------------------------------
__global__ __launch_bounds__(512) void bq_attn_out(const u16* __restrict__ hc,
                                                   const float* __restrict__ Mu,
                                                   const float* __restrict__ outW,
                                                   const float* __restrict__ outb,
                                                   float* __restrict__ out) {
    __shared__ float sc[BQ_S];
    __shared__ float red[512];
    const int b    = blockIdx.x;
    const int tid  = threadIdx.x;
    const int wv   = tid >> 5;
    const int lane = tid & 31;

    // Phase 1: scores. Wave handles rows t = wv, wv+16, ...; lane covers 16 chans.
    float mu[16];
#pragma unroll
    for (int i = 0; i < 16; i++) mu[i] = Mu[lane * 16 + i];
    for (int t = wv; t < BQ_S; t += 16) {
        const uint4* rv = (const uint4*)(hc + ((size_t)b * BQ_S + t) * 512 + lane * 16);
        uint4 p0 = rv[0], p1 = rv[1];
        u32 uu[8] = {p0.x, p0.y, p0.z, p0.w, p1.x, p1.y, p1.z, p1.w};
        float s = 0.f;
#pragma unroll
        for (int q = 0; q < 8; q++) {
            s += bf2f((u16)(uu[q] & 0xFFFFu)) * mu[2 * q];
            s += bf2f((u16)(uu[q] >> 16)) * mu[2 * q + 1];
        }
#pragma unroll
        for (int off = 16; off >= 1; off >>= 1) s += __shfl_xor(s, off, 32);
        if (lane == 0) sc[t] = s;
    }
    __syncthreads();

    // Phase 2: softmax over S (in LDS)
    float lm = -1e30f;
    for (int t = tid; t < BQ_S; t += 512) lm = fmaxf(lm, sc[t]);
    red[tid] = lm; __syncthreads();
    for (int s2 = 256; s2 > 0; s2 >>= 1) {
        if (tid < s2) red[tid] = fmaxf(red[tid], red[tid + s2]);
        __syncthreads();
    }
    float mx = red[0]; __syncthreads();
    float ls = 0.f;
    for (int t = tid; t < BQ_S; t += 512) { float e = __expf(sc[t] - mx); sc[t] = e; ls += e; }
    red[tid] = ls; __syncthreads();
    for (int s2 = 256; s2 > 0; s2 >>= 1) {
        if (tid < s2) red[tid] += red[tid + s2];
        __syncthreads();
    }
    float inv = 1.f / red[0]; __syncthreads();

    // Phase 3: ctx[c] (c = tid) and final projection
    float accv = 0.f;
    for (int t = 0; t < BQ_S; t++)
        accv += sc[t] * bf2f(hc[((size_t)b * BQ_S + t) * 512 + tid]);
    accv *= inv;
    red[tid] = accv * outW[tid]; __syncthreads();
    for (int s2 = 256; s2 > 0; s2 >>= 1) {
        if (tid < s2) red[tid] += red[tid + s2];
        __syncthreads();
    }
    if (tid == 0) out[b] = red[0] + outb[0];
}

// ---------------------------------------------------------------------------
// Workspace layout (bytes, 256-aligned):
//   Wc    :       327,680  (160*1024 bf16)
//   bias  :         4,096  (1024 f32)
//   x     :    41,943,040  (131072*160 bf16)
//   G     :   268,435,456  (131072*1024 bf16)
//   hc    :   134,217,728  (131072*512 bf16)
//   total ~= 445 MB
// ---------------------------------------------------------------------------
extern "C" void kernel_launch(void* const* d_in, const int* in_sizes, int n_in,
                              void* d_out, int out_size, void* d_ws, size_t ws_size,
                              hipStream_t stream) {
    const float* X    = (const float*)d_in[0];
    const float* emb  = (const float*)d_in[1];
    const float* nW   = (const float*)d_in[2];
    const float* nb   = (const float*)d_in[3];
    const float* Wf   = (const float*)d_in[4];
    const float* bfv  = (const float*)d_in[5];
    const float* Wb   = (const float*)d_in[6];
    const float* bbv  = (const float*)d_in[7];
    const float* Mu   = (const float*)d_in[8];
    const float* outW = (const float*)d_in[9];
    const float* outb = (const float*)d_in[10];

    char* ws = (char*)d_ws;
    u16*   Wc   = (u16*)ws;
    float* bias = (float*)(ws + 327680);
    u16*   xq   = (u16*)(ws + 331776);
    u16*   G    = (u16*)(ws + 331776ull + 41943040ull);
    u16*   hc   = (u16*)(ws + 331776ull + 41943040ull + 268435456ull);

    bq_prep_w<<<(BQ_KP * BQ_N + 255) / 256, 256, 0, stream>>>(Wf, Wb, bfv, bbv, Wc, bias);
    bq_build_x<<<BQ_M / 8, 256, 0, stream>>>(X, emb, nW, nb, xq);
    bq_gemm_gates<<<dim3(BQ_M / 128, BQ_N / 64), 256, 0, stream>>>(xq, Wc, bias, G);
    bq_fo_pool<<<32768 / 256, 256, 0, stream>>>(G, hc);
    bq_attn_out<<<BQ_B, 512, 0, stream>>>(hc, Mu, outW, outb, (float*)d_out);
}
